// MemoryEfficientAttention_8701603741986
// MI455X (gfx1250) — matrix operand order repaired and checked by reading, not compiled, not run on hardware
//
#include <hip/hip_runtime.h>

// ---------------------------------------------------------------------------
// Flash-attention forward for MI455X (gfx1250, wave32, WMMA).
// B=2, S=4096, H=16, D=128, fp32 in/out. Query-side mask folded into Q
// (masked row -> q=0 -> scores=0 -> uniform softmax -> mean(V), matching ref).
// ---------------------------------------------------------------------------

typedef __attribute__((ext_vector_type(16))) __bf16 v16bf;
typedef __attribute__((ext_vector_type(8)))  __bf16 v8bf;
typedef __attribute__((ext_vector_type(2)))  __bf16 v2bf;
typedef __attribute__((ext_vector_type(8)))  float  v8f;

#define S_LEN   4096
#define NHEAD   16
#define DHEAD   128
#define QTILE   128      // query rows per workgroup (8 waves x 16)
#define KTILE   64       // key rows staged in LDS per iteration

// LDS strides in bytes (16B aligned, padded to dodge bank conflicts)
#define KSTR    272      // 128 bf16 + 8 pad
#define VTSTR   144      // 64 bf16 + 8 pad  (VT[d][k])
#define PSTR    144      // 64 bf16 + 8 pad  (P[row][k], per-wave patch)

static __device__ __forceinline__ __bf16 tobf(float x) { return (__bf16)x; }

// contiguous 16 bf16 (B-matrix fragment): element i <-> K = base + i
static __device__ __forceinline__ v16bf ldB16(const __bf16* p) {
  v8bf lo = *(const v8bf*)p;
  v8bf hi = *(const v8bf*)(p + 8);
  v16bf r;
#pragma unroll
  for (int i = 0; i < 8; ++i) { r[i] = lo[i]; r[i + 8] = hi[i]; }
  return r;
}

// A-matrix fragment halves: elements 0..7 = p[0..7], elements 8..15 = p[16..23]
static __device__ __forceinline__ v16bf ldA16(const __bf16* p) {
  v8bf lo = *(const v8bf*)p;
  v8bf hi = *(const v8bf*)(p + 16);
  v16bf r;
#pragma unroll
  for (int i = 0; i < 8; ++i) { r[i] = lo[i]; r[i + 8] = hi[i]; }
  return r;
}

static __device__ __forceinline__ float hmax16(float v) {
#pragma unroll
  for (int m = 1; m <= 8; m <<= 1) v = fmaxf(v, __shfl_xor(v, m, 32));
  return v;
}
static __device__ __forceinline__ float hsum16(float v) {
#pragma unroll
  for (int m = 1; m <= 8; m <<= 1) v += __shfl_xor(v, m, 32);
  return v;
}

__global__ __launch_bounds__(256, 1)
void MemoryEfficientAttention_8701603741986_kernel(
    const float* __restrict__ Q, const float* __restrict__ K,
    const float* __restrict__ V, const unsigned char* __restrict__ MSK,
    float* __restrict__ O) {
  __shared__ __align__(16) char ldsK[KTILE * KSTR];     // K tile, bf16 row-major
  __shared__ __align__(16) char ldsVT[DHEAD * VTSTR];   // V tile, bf16 transposed
  __shared__ __align__(16) char ldsP[QTILE * PSTR];     // per-wave P staging

  const float scale = 0.08838834764831845f;  // 1/sqrt(128)

  const int wg  = blockIdx.x;
  const int qt  = wg & 31;          // query tile index (S/QTILE = 32)
  const int h   = (wg >> 5) & 15;
  const int b   = wg >> 9;
  const int tid = threadIdx.x;
  const int lane = tid & 31;
  const int wave = tid >> 5;
  const int lr   = lane & 15;
  const int h8   = (lane < 16) ? 0 : 8;   // A-fragment / C-row half offset
  const int h16  = h8 * 2;                // unused; B fragments also use h8

  // ---- load Q tile as 4 A-fragments (mask & scale folded in) ----------------
  const int qrow = qt * QTILE + wave * 16 + lr;  // global query row (lane's A row)
  const float* qp =
      Q + (((size_t)b * S_LEN + qrow) * NHEAD + h) * DHEAD;
  const float qm = MSK[(size_t)b * S_LEN + qrow] ? scale : 0.0f;

  v16bf qa[4];
#pragma unroll
  for (int f = 0; f < 4; ++f) {
    const int d0 = 32 * f + h8;
    float4 x0 = *(const float4*)(qp + d0);
    float4 x1 = *(const float4*)(qp + d0 + 4);
    float4 x2 = *(const float4*)(qp + d0 + 16);
    float4 x3 = *(const float4*)(qp + d0 + 20);
    qa[f][0]  = tobf(x0.x * qm); qa[f][1]  = tobf(x0.y * qm);
    qa[f][2]  = tobf(x0.z * qm); qa[f][3]  = tobf(x0.w * qm);
    qa[f][4]  = tobf(x1.x * qm); qa[f][5]  = tobf(x1.y * qm);
    qa[f][6]  = tobf(x1.z * qm); qa[f][7]  = tobf(x1.w * qm);
    qa[f][8]  = tobf(x2.x * qm); qa[f][9]  = tobf(x2.y * qm);
    qa[f][10] = tobf(x2.z * qm); qa[f][11] = tobf(x2.w * qm);
    qa[f][12] = tobf(x3.x * qm); qa[f][13] = tobf(x3.y * qm);
    qa[f][14] = tobf(x3.z * qm); qa[f][15] = tobf(x3.w * qm);
  }

  // ---- flash state ----------------------------------------------------------
  v8f o[8];
#pragma unroll
  for (int j = 0; j < 8; ++j) o[j] = (v8f)0.0f;
  float rm[8], rs[8];
#pragma unroll
  for (int r = 0; r < 8; ++r) { rm[r] = -__builtin_huge_valf(); rs[r] = 0.0f; }

  // ---- main loop over 64-key tiles ------------------------------------------
  for (int kb = 0; kb < S_LEN; kb += KTILE) {
    __syncthreads();  // previous tile's compute done before overwrite

    // K tile -> LDS (bf16 row-major). 256 thr: 4 threads/row, 32 f32 each.
    {
      const int krow = tid >> 2;
      const int dseg = (tid & 3) << 5;
      const float* kp =
          K + (((size_t)b * S_LEN + kb + krow) * NHEAD + h) * DHEAD + dseg;
      __bf16* dk = (__bf16*)(ldsK + krow * KSTR) + dseg;
#pragma unroll
      for (int g = 0; g < 4; ++g) {
        float4 x0 = *(const float4*)(kp + 8 * g);
        float4 x1 = *(const float4*)(kp + 8 * g + 4);
        v8bf y;
        y[0] = tobf(x0.x); y[1] = tobf(x0.y); y[2] = tobf(x0.z); y[3] = tobf(x0.w);
        y[4] = tobf(x1.x); y[5] = tobf(x1.y); y[6] = tobf(x1.z); y[7] = tobf(x1.w);
        *(v8bf*)(dk + 8 * g) = y;
      }
      if (kb + KTILE < S_LEN)
        __builtin_prefetch(kp + (size_t)KTILE * NHEAD * DHEAD, 0, 1);  // global_prefetch_b8
    }

    // V tile -> LDS transposed (VT[d][k], bf16). 512 combos of (key-pair, dseg8).
#pragma unroll
    for (int c = tid; c < 512; c += 256) {
      const int p2 = c >> 4;          // key pair 0..31
      const int d0 = (c & 15) << 3;   // d segment base
      const float* vp =
          V + (((size_t)b * S_LEN + kb + 2 * p2) * NHEAD + h) * DHEAD + d0;
      float4 a0 = *(const float4*)(vp);
      float4 a1 = *(const float4*)(vp + 4);
      float4 b0 = *(const float4*)(vp + NHEAD * DHEAD);
      float4 b1 = *(const float4*)(vp + NHEAD * DHEAD + 4);
      float av[8] = {a0.x, a0.y, a0.z, a0.w, a1.x, a1.y, a1.z, a1.w};
      float bv[8] = {b0.x, b0.y, b0.z, b0.w, b1.x, b1.y, b1.z, b1.w};
      char* base = ldsVT + (size_t)d0 * VTSTR + 4 * p2;
#pragma unroll
      for (int i = 0; i < 8; ++i) {
        v2bf pk; pk[0] = tobf(av[i]); pk[1] = tobf(bv[i]);
        *(v2bf*)(base + i * VTSTR) = pk;
      }
      if (kb + KTILE < S_LEN)
        __builtin_prefetch(vp + (size_t)KTILE * NHEAD * DHEAD, 0, 1);
    }

    __syncthreads();

    // ---- scores: 4x (16q x 16k) tiles, each 4 WMMA over D=128 --------------
    float s[4][8];
#pragma unroll
    for (int kt = 0; kt < 4; ++kt) {
      const __bf16* kr = (const __bf16*)(ldsK + (kt * 16 + lr) * KSTR);
      v8f c = (v8f)0.0f;
#pragma unroll
      for (int f = 0; f < 4; ++f) {
        v16bf bk = ldA16(kr + 32 * f + h8);
        c = __builtin_amdgcn_wmma_f32_16x16x32_bf16(
                false, qa[f], false, bk, (short)0, c, false, false);
      }
#pragma unroll
      for (int r = 0; r < 8; ++r) s[kt][r] = c[r];
    }

    // ---- online softmax over the 64-key block ------------------------------
#pragma unroll
    for (int r = 0; r < 8; ++r) {
      float tm = fmaxf(fmaxf(s[0][r], s[1][r]), fmaxf(s[2][r], s[3][r]));
      tm = hmax16(tm);
      const float mn   = fmaxf(rm[r], tm);
      const float corr = __expf(rm[r] - mn);
      rm[r] = mn;
      float ps = 0.0f;
      char* prw = ldsP + ((size_t)(wave * 16 + r + h8)) * PSTR;
#pragma unroll
      for (int kt = 0; kt < 4; ++kt) {
        float p = __expf(s[kt][r] - mn);
        ps += p;
        *(__bf16*)(prw + (kt * 16 + lr) * 2) = tobf(p);  // C-layout -> LDS
      }
      rs[r] = rs[r] * corr + hsum16(ps);
#pragma unroll
      for (int j = 0; j < 8; ++j) o[j][r] *= corr;
    }

    // ---- P*V: reload P as A-fragments (wave-private LDS, in-order DS) ------
    const __bf16* prr = (const __bf16*)(ldsP + (size_t)(wave * 16 + lr) * PSTR);
    v16bf pa[2];
#pragma unroll
    for (int f = 0; f < 2; ++f) pa[f] = ldA16(prr + 32 * f + h8);
#pragma unroll
    for (int j = 0; j < 8; ++j) {
      const __bf16* vtr = (const __bf16*)(ldsVT + (size_t)(16 * j + lr) * VTSTR);
#pragma unroll
      for (int f = 0; f < 2; ++f) {
        v16bf bv = ldA16(vtr + 32 * f + h8);
        o[j] = __builtin_amdgcn_wmma_f32_16x16x32_bf16(
                   false, pa[f], false, bv, (short)0, o[j], false, false);
      }
    }
  }

  // ---- epilogue: normalize and store ---------------------------------------
  float inv[8];
#pragma unroll
  for (int r = 0; r < 8; ++r) inv[r] = 1.0f / rs[r];
#pragma unroll
  for (int r = 0; r < 8; ++r) {
    const int row = qt * QTILE + wave * 16 + r + h8;   // C-layout row
    float* op = O + (((size_t)b * S_LEN + row) * NHEAD + h) * DHEAD + lr;
#pragma unroll
    for (int j = 0; j < 8; ++j) op[16 * j] = o[j][r] * inv[r];
  }
}

extern "C" void kernel_launch(void* const* d_in, const int* in_sizes, int n_in,
                              void* d_out, int out_size, void* d_ws, size_t ws_size,
                              hipStream_t stream) {
  (void)in_sizes; (void)n_in; (void)d_ws; (void)ws_size; (void)out_size;
  const float* Q = (const float*)d_in[0];
  const float* K = (const float*)d_in[1];
  const float* V = (const float*)d_in[2];
  const unsigned char* M = (const unsigned char*)d_in[3];
  float* O = (float*)d_out;
  // grid: B(2) * H(16) * S/QTILE(32) = 1024 workgroups, 256 threads (8 waves)
  dim3 grid(1024), block(256);
  hipLaunchKernelGGL(MemoryEfficientAttention_8701603741986_kernel, grid, block,
                     0, stream, Q, K, V, M, O);
}
